// ATT_learner_68796786147775
// MI455X (gfx1250) — compile-verified, matrix operand order checked by reading, not run on hardware
//
#include <hip/hip_runtime.h>

typedef __attribute__((ext_vector_type(16))) _Float16 v16h;
typedef __attribute__((ext_vector_type(8)))  _Float16 v8h;
typedef __attribute__((ext_vector_type(8)))  float    v8f;
typedef int v4i __attribute__((vector_size(16)));   // generic GCC vector, matches builtin proto

#define NNODES 12288
#define DFEAT  256
#define KTOP   31
#define BK     128          // block tile edge (rows == cols)
#define LDSH   264          // padded halves per node (256 + 8) -> 528B stride, bank-conflict free

// CDNA5 async global->LDS path (ASYNCcnt-tracked), with safe fallback.
#if __has_builtin(__builtin_amdgcn_global_load_async_to_lds_b128) && \
    __has_builtin(__builtin_amdgcn_s_wait_asynccnt)
#define USE_ASYNC_LDS 1
#endif

__device__ __forceinline__ void stage16(const _Float16* __restrict__ g,
                                        _Float16* __restrict__ l) {
#ifdef USE_ASYNC_LDS
  __builtin_amdgcn_global_load_async_to_lds_b128(
      (__attribute__((address_space(1))) v4i*)(void*)g,
      (__attribute__((address_space(3))) v4i*)(void*)l,
      /*offset=*/0, /*cpol=*/0);
#else
  *(v8h*)l = *(const v8h*)g;
#endif
}

// ---------------------------------------------------------------------------
// Kernel 1: h = L2normalize(relu(x*w0)*w1) per row, stored as f16 in d_ws.
// One block per node, 256 threads = one feature each (D == 256).
// ---------------------------------------------------------------------------
__global__ __launch_bounds__(256) void normalize_rows_f16(
    const float* __restrict__ x, const float* __restrict__ w0,
    const float* __restrict__ w1, _Float16* __restrict__ h) {
  __shared__ float red[8];
  const int n = blockIdx.x;
  const int d = threadIdx.x;
  float v = x[(size_t)n * DFEAT + d] * w0[d];
  v = fmaxf(v, 0.0f) * w1[d];
  float s = v * v;
  // wave32 butterfly reduction
  #pragma unroll
  for (int off = 16; off > 0; off >>= 1) s += __shfl_xor(s, off, 32);
  const int wave = d >> 5, lane = d & 31;
  if (lane == 0) red[wave] = s;
  __syncthreads();
  if (d == 0) {
    float t = 0.0f;
    #pragma unroll
    for (int i = 0; i < 8; ++i) t += red[i];
    red[0] = 1.0f / fmaxf(sqrtf(t), 1e-12f);
  }
  __syncthreads();
  h[(size_t)n * DFEAT + d] = (_Float16)(v * red[0]);
}

// ---------------------------------------------------------------------------
// Kernel 2: S = H * H^T via v_wmma_f32_16x16x32_f16.
// Block tile 128x128, 8 waves: wave w owns the 16-row strip, iterates 8
// column tiles against an LDS-staged 128-node column panel (staged via
// GLOBAL_LOAD_ASYNC_TO_LDS_B128 when available).
//
// Fragment gather for a 16x32 f16 A tile (per ISA 7.12.2): lane L (L<16)
// holds row M=L, halves K = {0..7, 16..23}; lane L+16 holds row M=L,
// halves K = {8..15, 24..31}. Two 16B chunks per fragment. B (=H^T tile)
// uses the transpose-symmetric layout -> same gather keyed by column node.
// ---------------------------------------------------------------------------
__global__ __launch_bounds__(256) void gemm_sim(
    const _Float16* __restrict__ H, float* __restrict__ S) {
  __shared__ _Float16 Bs[BK * LDSH];   // 67,584 bytes

  const int tid  = threadIdx.x;
  const int lane = tid & 31;
  const int wave = tid >> 5;
  const int rowBase = blockIdx.y * BK;
  const int colBase = blockIdx.x * BK;

  // Stage column panel: 128 nodes x 256 halves = 4096 16B chunks, 16/thread.
  #pragma unroll
  for (int i = 0; i < 16; ++i) {
    const int c    = tid + i * 256;     // chunk id
    const int node = c >> 5;            // /32 chunks per node
    const int kc   = (c & 31) << 3;     // half offset
    stage16(&H[(size_t)(colBase + node) * DFEAT + kc], &Bs[node * LDSH + kc]);
  }
#ifdef USE_ASYNC_LDS
  __builtin_amdgcn_s_wait_asynccnt(0); // my async copies landed in LDS
#endif
  __syncthreads();                     // everyone's copies visible

  const int m    = lane & 15;           // outer index within tile
  const int koff = (lane >> 4) << 3;    // 0 or 8 halves

  // A fragments for this wave's 16 rows, entire K=256 (8 fragments), held
  // in registers and reused across all 8 column tiles.
  v16h afrag[8];
  const size_t arow = (size_t)(rowBase + wave * 16 + m) * DFEAT;
  #pragma unroll
  for (int k = 0; k < 8; ++k) {
    v8h lo = *(const v8h*)&H[arow + k * 32 + koff];
    v8h hi = *(const v8h*)&H[arow + k * 32 + 16 + koff];
    afrag[k] = __builtin_shufflevector(lo, hi, 0, 1, 2, 3, 4, 5, 6, 7,
                                               8, 9, 10, 11, 12, 13, 14, 15);
  }

  #pragma unroll
  for (int j = 0; j < 8; ++j) {
    v8f acc = {};
    const int cnode = j * 16 + m;       // node within LDS panel
    #pragma unroll
    for (int k = 0; k < 8; ++k) {
      v8h lo = *(const v8h*)&Bs[cnode * LDSH + k * 32 + koff];
      v8h hi = *(const v8h*)&Bs[cnode * LDSH + k * 32 + 16 + koff];
      v16h bfrag = __builtin_shufflevector(lo, hi, 0, 1, 2, 3, 4, 5, 6, 7,
                                                   8, 9, 10, 11, 12, 13, 14, 15);
      acc = __builtin_amdgcn_wmma_f32_16x16x32_f16(
          /*neg_a=*/false, afrag[k], /*neg_b=*/false, bfrag,
          /*c_mod=*/(short)0, acc, /*reuse_a=*/false, /*reuse_b=*/false);
    }
    // C/D layout: VGPR r -> (M = r + 8*(lane>=16), N = lane&15)
    const int col   = colBase + j * 16 + m;
    const int rbase = rowBase + wave * 16 + ((lane >> 4) << 3);
    #pragma unroll
    for (int r = 0; r < 8; ++r)
      S[(size_t)(rbase + r) * NNODES + col] = acc[r];
  }
}

// ---------------------------------------------------------------------------
// Kernel 3: per-row top-K(31) threshold via bisection on an LDS-cached row,
// then out = relu(s * mask) == (s >= thr && s > 0) ? s : 0, in place.
// ---------------------------------------------------------------------------
__global__ __launch_bounds__(256) void topk_mask(float* __restrict__ S) {
  __shared__ float row[NNODES];        // 48 KB
  __shared__ int   cnt;
  const int n   = blockIdx.x;
  const int tid = threadIdx.x;
  float* __restrict__ srow = S + (size_t)n * NNODES;

  for (int i = tid; i < NNODES; i += 256) row[i] = srow[i];
  __syncthreads();

  float lo = -2.0f, hi = 2.0f;         // cosine similarities are in [-1, 1]
  for (int it = 0; it < 30; ++it) {
    const float mid = 0.5f * (lo + hi);
    if (tid == 0) cnt = 0;
    __syncthreads();
    int c = 0;
    for (int i = tid; i < NNODES; i += 256) c += (row[i] >= mid) ? 1 : 0;
    #pragma unroll
    for (int off = 16; off > 0; off >>= 1) c += __shfl_xor(c, off, 32);
    if ((tid & 31) == 0) atomicAdd(&cnt, c);
    __syncthreads();
    if (cnt >= KTOP) lo = mid; else hi = mid;
    __syncthreads();
  }
  const float thr = lo;
  for (int i = tid; i < NNODES; i += 256) {
    const float v = row[i];
    srow[i] = (v >= thr && v > 0.0f) ? v : 0.0f;
  }
}

// ---------------------------------------------------------------------------
extern "C" void kernel_launch(void* const* d_in, const int* in_sizes, int n_in,
                              void* d_out, int out_size, void* d_ws, size_t ws_size,
                              hipStream_t stream) {
  const float* x  = (const float*)d_in[0];
  const float* w0 = (const float*)d_in[1];
  const float* w1 = (const float*)d_in[2];
  float*       S  = (float*)d_out;
  _Float16*    H  = (_Float16*)d_ws;   // needs N*D*2 = 6.3 MB of scratch

  normalize_rows_f16<<<NNODES, 256, 0, stream>>>(x, w0, w1, H);

  dim3 grid(NNODES / BK, NNODES / BK); // 96 x 96 blocks
  gemm_sim<<<grid, 256, 0, stream>>>(H, S);

  topk_mask<<<NNODES, 256, 0, stream>>>(S);
}
